// SelfAttention_76708115906891
// MI455X (gfx1250) — matrix operand order repaired and checked by reading, not compiled, not run on hardware
//
#include <hip/hip_runtime.h>
#include <hip/hip_bf16.h>

#define NPIX 4096
#define CDIM 256
#define NH   4
#define HD   64
#define BATCH 4

typedef __bf16 bf16;
typedef __attribute__((ext_vector_type(16))) __bf16 bf16x16;
typedef __attribute__((ext_vector_type(8)))  __bf16 bf16x8;
typedef __attribute__((ext_vector_type(8)))  float  f32x8;
typedef __attribute__((ext_vector_type(4)))  unsigned u32x4;
typedef __attribute__((ext_vector_type(8)))  unsigned u32x8;

union Frag { bf16x16 v; bf16x8 h[2]; };

__device__ __forceinline__ f32x8 wmma_bf16(bf16x16 a, bf16x16 b, f32x8 c) {
  // D = A(16x32 bf16) * B(32x16 bf16) + C(16x16 f32)
  return __builtin_amdgcn_wmma_f32_16x16x32_bf16(false, a, false, b, (short)0, c, false, false);
}

__device__ __forceinline__ f32x8 fzero() {
  f32x8 z = {0.f,0.f,0.f,0.f,0.f,0.f,0.f,0.f};
  return z;
}

// ---------------------------------------------------------------------------
// CDNA5 Tensor Data Mover: one-instruction DMA of a 2D bf16 tile
// (nrows x row_len, global row stride = row_stride_elems) into contiguous LDS.
// Descriptor layout per CDNA5 ISA (groups 0+1, 2D form).
// Tracked by TENSORcnt; issued by one wave, EXEC-independent.
// ---------------------------------------------------------------------------
__device__ __forceinline__ void tdm_load_2d(uint32_t lds_addr, const void* gaddr,
                                            uint32_t row_len_elems, uint32_t nrows,
                                            uint32_t row_stride_elems) {
  uint64_t ga = (uint64_t)(uintptr_t)gaddr;
  u32x4 g0;
  g0[0] = 1u;                                   // count=1, is_restore=0, gather=0
  g0[1] = lds_addr;                             // lds_addr [63:32]
  g0[2] = (uint32_t)ga;                         // global_addr low
  g0[3] = (uint32_t)(ga >> 32) | 0x80000000u;   // global_addr hi | type=2 ([127:126])
  const uint32_t TD = 1u << 20;                 // generous tensor dims (no OOB clip)
  u32x8 g1;
  g1[0] = 0x00010000u;                          // workgroup_mask=0, data_size=1 (2B)
  g1[1] = (TD & 0xFFFFu) << 16;                 // tensor_dim0 low16 -> [47:32]=abar=0
  g1[2] = (TD >> 16) | ((TD & 0xFFFFu) << 16);  // tensor_dim0 hi | tensor_dim1 low
  g1[3] = (TD >> 16) | (row_len_elems << 16);   // tensor_dim1 hi | tile_dim0
  g1[4] = nrows;                                // tile_dim1 (tile_dim2 = 0)
  g1[5] = row_stride_elems;                     // tensor_dim0_stride low32
  g1[6] = 0u;                                   // stride hi | dim1_stride low
  g1[7] = 0u;                                   // dim1_stride hi
  asm volatile("tensor_load_to_lds %0, %1" :: "s"(g0), "s"(g1) : "memory");
}

// ---------------------------------------------------------------------------
// Kernel 1a: x [B][C][N] f32  ->  xT [B][N][C] bf16
// ---------------------------------------------------------------------------
__global__ void convert_x_kernel(const float* __restrict__ x, bf16* __restrict__ xT) {
  size_t idx = (size_t)blockIdx.x * blockDim.x + threadIdx.x;
  const size_t per = (size_t)CDIM * NPIX;
  if (idx >= (size_t)BATCH * per) return;
  int    bb = (int)(idx / per);
  size_t rr = idx % per;
  int n = (int)(rr / CDIM);
  int c = (int)(rr % CDIM);
  xT[idx] = (bf16)x[(size_t)bb * per + (size_t)c * NPIX + n];
}

// ---------------------------------------------------------------------------
// Kernel 1b: wq|wk|wv|wo [256][256] f32 -> contiguous bf16 blocks
// ---------------------------------------------------------------------------
__global__ void convert_w_kernel(const float* __restrict__ wq, const float* __restrict__ wk,
                                 const float* __restrict__ wv, const float* __restrict__ wo,
                                 bf16* __restrict__ wbf) {
  int idx = blockIdx.x * blockDim.x + threadIdx.x;   // 4*65536
  int m = idx >> 16;
  int i = idx & 65535;
  const float* w = (m == 0) ? wq : (m == 1) ? wk : (m == 2) ? wv : wo;
  wbf[idx] = (bf16)w[i];
}

// ---------------------------------------------------------------------------
// Kernel 2: QKV projection, register-double-buffered WMMA GEMM.
// Each wave computes a 16(d) x 64(n) tile over C=256 in K=32 chunks.
//   q -> [B][N][C] bf16 (pre-scaled by hd^-0.5), k -> [B][N][C], v -> [B][C][N]
// ---------------------------------------------------------------------------
__global__ __launch_bounds__(256) void proj_qkv_kernel(
    const bf16* __restrict__ xT, const bf16* __restrict__ wbf,
    const float* __restrict__ biasq, const float* __restrict__ biask,
    const float* __restrict__ biasv,
    bf16* __restrict__ qb, bf16* __restrict__ kb, bf16* __restrict__ vtb) {
  const int wave = threadIdx.x >> 5;
  const int lane = threadIdx.x & 31;
  const int lmod = lane & 15;
  const int h2   = lane >> 4;

  int wid  = blockIdx.x * 8 + wave;     // 3 * 4 * 16 * 64 = 12288 waves
  int proj = wid >> 12;                 // 0:q 1:k 2:v
  int rem  = wid & 4095;
  int bb   = rem >> 10;
  int rem2 = rem & 1023;
  int dbase = (rem2 >> 6) * 16;
  int nbase = (rem2 & 63) * 64;

  const bf16*  w    = wbf + (size_t)proj * 65536;
  const float* bias = (proj == 0) ? biasq : (proj == 1) ? biask : biasv;

  f32x8 acc[4];
#pragma unroll
  for (int t = 0; t < 4; t++) acc[t] = fzero();

  const bf16* wrow = w + (size_t)(dbase + lmod) * CDIM;
  const bf16* xrow = xT + ((size_t)bb * NPIX + nbase + lmod) * CDIM;

  Frag a_cur, a_nxt, b_cur[4], b_nxt[4];
  a_cur.h[0] = *(const bf16x8*)(wrow + 16 * h2);
  a_cur.h[1] = *(const bf16x8*)(wrow + 16 * h2 + 8);
#pragma unroll
  for (int t = 0; t < 4; t++) {
    const bf16* xp = xrow + (size_t)(16 * t) * CDIM + 16 * h2;
    b_cur[t].h[0] = *(const bf16x8*)(xp);
    b_cur[t].h[1] = *(const bf16x8*)(xp + 8);
  }

#pragma unroll
  for (int kc = 0; kc < 8; kc++) {
    if (kc < 7) {                      // prefetch next K-chunk while MMAs run
      const int c1 = (kc + 1) * 32;
      a_nxt.h[0] = *(const bf16x8*)(wrow + c1 + 16 * h2);
      a_nxt.h[1] = *(const bf16x8*)(wrow + c1 + 16 * h2 + 8);
#pragma unroll
      for (int t = 0; t < 4; t++) {
        const bf16* xp = xrow + (size_t)(16 * t) * CDIM + c1 + 16 * h2;
        b_nxt[t].h[0] = *(const bf16x8*)(xp);
        b_nxt[t].h[1] = *(const bf16x8*)(xp + 8);
      }
    }
#pragma unroll
    for (int t = 0; t < 4; t++) acc[t] = wmma_bf16(a_cur.v, b_cur[t].v, acc[t]);
    a_cur = a_nxt;
#pragma unroll
    for (int t = 0; t < 4; t++) b_cur[t] = b_nxt[t];
  }

  f32x8 bv8 = *(const f32x8*)(bias + dbase + 8 * h2);   // bias for d = dbase+8*h2+r
  const float scale = (proj == 0) ? 0.125f : 1.0f;      // hd^-0.5 folded into q

  if (proj < 2) {
    bf16* outb = (proj == 0) ? qb : kb;
#pragma unroll
    for (int t = 0; t < 4; t++) {
      bf16x8 pk;
#pragma unroll
      for (int r = 0; r < 8; r++) pk[r] = (bf16)((acc[t][r] + bv8[r]) * scale);
      int n = nbase + 16 * t + lmod;
      *(bf16x8*)(outb + ((size_t)bb * NPIX + n) * CDIM + dbase + 8 * h2) = pk;
    }
  } else {
#pragma unroll
    for (int t = 0; t < 4; t++) {
      int n = nbase + 16 * t + lmod;
#pragma unroll
      for (int r = 0; r < 8; r++) {
        int d = dbase + 8 * h2 + r;
        vtb[((size_t)bb * CDIM + d) * NPIX + n] = (bf16)(acc[t][r] + bv8[r]);
      }
    }
  }
}

// ---------------------------------------------------------------------------
// Kernel 3: flash attention.
// Workgroup = 8 waves = 128 q rows of one (b, head). K/V tiles for 64-key
// blocks are staged into double-buffered LDS by the Tensor Data Mover
// (tensor_load_to_lds + s_wait_tensorcnt): wave 0 issues one descriptor DMA
// per 8KB tile, so K/V L2 traffic is paid once per workgroup (8x cut) with
// near-zero issue overhead. Per block: 8 WMMA for S, online softmax,
// P via per-wave LDS relayout, 8 WMMA for P.V.
// ---------------------------------------------------------------------------
__global__ __launch_bounds__(256) void attn_kernel(
    const bf16* __restrict__ qb, const bf16* __restrict__ kb,
    const bf16* __restrict__ vtb, bf16* __restrict__ aob) {
  __shared__ __align__(16) bf16 ldsK[2][64 * 64];   // [key][d]   8KB x2
  __shared__ __align__(16) bf16 ldsV[2][64 * 64];   // [d][key]   8KB x2
  __shared__ __align__(16) bf16 ldsP[8][16 * 64];   // per-wave P 2KB x8

  const int tid  = threadIdx.x;
  const int wave = tid >> 5;
  const int lane = tid & 31;
  const int lmod = lane & 15;
  const int h2   = lane >> 4;

  // 512 blocks: 4 batches x 4 heads x 32 q-blocks of 128 rows
  int bid    = blockIdx.x;
  int bb     = bid >> 7;
  int rem    = bid & 127;
  int hh     = rem >> 5;
  int qbase0 = (rem & 31) * 128;
  int qbase  = qbase0 + wave * 16;

  const bf16* khead = kb + (size_t)bb * NPIX * CDIM + hh * HD;
  const bf16* vhead = vtb + ((size_t)bb * CDIM + hh * HD) * NPIX;
  bf16* ldsw = &ldsP[wave][0];

  // q A-fragments (K chunks 0..31 and 32..63), resident for the whole loop
  const bf16* qrow = qb + ((size_t)bb * NPIX + qbase + lmod) * CDIM + hh * HD;
  Frag qa0, qa1;
  qa0.h[0] = *(const bf16x8*)(qrow + 16 * h2);
  qa0.h[1] = *(const bf16x8*)(qrow + 16 * h2 + 8);
  qa1.h[0] = *(const bf16x8*)(qrow + 32 + 16 * h2);
  qa1.h[1] = *(const bf16x8*)(qrow + 40 + 16 * h2);

  float m8[8], l8[8];
#pragma unroll
  for (int r = 0; r < 8; r++) { m8[r] = -3.0e38f; l8[r] = 0.0f; }
  f32x8 o[4];
#pragma unroll
  for (int t = 0; t < 4; t++) o[t] = fzero();

  // stage one 64-key block via TDM (wave 0 only): K tile + V tile, 8KB each
  auto stage = [&](int buf, int kb0) {
    tdm_load_2d((uint32_t)(uintptr_t)&ldsK[buf][0],
                khead + (size_t)kb0 * CDIM, 64, 64, CDIM);
    tdm_load_2d((uint32_t)(uintptr_t)&ldsV[buf][0],
                vhead + kb0, 64, 64, NPIX);
  };

  const int NB = NPIX / 64;        // 64 key blocks
  if (wave == 0) {
    stage(0, 0);
    stage(1, 64);
  }

  for (int j = 0; j < NB; j++) {
    const int buf = j & 1;
    const bf16* lk = &ldsK[buf][0];
    const bf16* lv = &ldsV[buf][0];

    if (wave == 0) {
      if (j + 1 < NB) __builtin_amdgcn_s_wait_tensorcnt(2);  // block j landed
      else            __builtin_amdgcn_s_wait_tensorcnt(0);
    }
    __syncthreads();

    // ---- S = q.k^T : four 16x16 score tiles over 64 keys --------------
    f32x8 s[4];
#pragma unroll
    for (int t = 0; t < 4; t++) {
      s[t] = fzero();
      const bf16* kp = lk + (16 * t + lmod) * 64 + 16 * h2;
      Frag fb;
      fb.h[0] = *(const bf16x8*)(kp);
      fb.h[1] = *(const bf16x8*)(kp + 8);
      s[t] = wmma_bf16(qa0.v, fb.v, s[t]);
      fb.h[0] = *(const bf16x8*)(kp + 32);
      fb.h[1] = *(const bf16x8*)(kp + 40);
      s[t] = wmma_bf16(qa1.v, fb.v, s[t]);
    }

    // ---- online softmax ------------------------------------------------
#pragma unroll
    for (int r = 0; r < 8; r++) {
      float t0 = fmaxf(fmaxf(s[0][r], s[1][r]), fmaxf(s[2][r], s[3][r]));
      t0 = fmaxf(t0, __shfl_xor(t0, 1, 32));
      t0 = fmaxf(t0, __shfl_xor(t0, 2, 32));
      t0 = fmaxf(t0, __shfl_xor(t0, 4, 32));
      t0 = fmaxf(t0, __shfl_xor(t0, 8, 32));
      float mn  = fmaxf(m8[r], t0);
      float al  = __expf(m8[r] - mn);
      float p0v = __expf(s[0][r] - mn);
      float p1v = __expf(s[1][r] - mn);
      float p2v = __expf(s[2][r] - mn);
      float p3v = __expf(s[3][r] - mn);
      float rs = (p0v + p1v) + (p2v + p3v);
      rs += __shfl_xor(rs, 1, 32);
      rs += __shfl_xor(rs, 2, 32);
      rs += __shfl_xor(rs, 4, 32);
      rs += __shfl_xor(rs, 8, 32);
      l8[r] = l8[r] * al + rs;
      m8[r] = mn;
      o[0][r] *= al; o[1][r] *= al; o[2][r] *= al; o[3][r] *= al;
      bf16* pw = ldsw + (r + 8 * h2) * 64 + lmod;
      pw[0]  = (bf16)p0v;
      pw[16] = (bf16)p1v;
      pw[32] = (bf16)p2v;
      pw[48] = (bf16)p3v;
    }

    // ---- reload P as two A fragments (16x64) ---------------------------
    Frag pa0, pa1;
    {
      const bf16* pp = ldsw + lmod * 64 + 16 * h2;
      pa0.h[0] = *(const bf16x8*)(pp);
      pa0.h[1] = *(const bf16x8*)(pp + 8);
      pa1.h[0] = *(const bf16x8*)(pp + 32);
      pa1.h[1] = *(const bf16x8*)(pp + 40);
    }

    // ---- O += P.V ------------------------------------------------------
#pragma unroll
    for (int t = 0; t < 4; t++) {
      const bf16* vp = lv + (16 * t + lmod) * 64 + 16 * h2;
      Frag fv;
      fv.h[0] = *(const bf16x8*)(vp);
      fv.h[1] = *(const bf16x8*)(vp + 8);
      o[t] = wmma_bf16(pa0.v, fv.v, o[t]);
      fv.h[0] = *(const bf16x8*)(vp + 32);
      fv.h[1] = *(const bf16x8*)(vp + 40);
      o[t] = wmma_bf16(pa1.v, fv.v, o[t]);
    }

    __syncthreads();                       // everyone done reading buf
    if (wave == 0 && j + 2 < NB) stage(buf, (j + 2) * 64);
  }

  // ---- normalize and store ao[b][n][c] (bf16, pixel-major) -------------
#pragma unroll
  for (int r = 0; r < 8; r++) m8[r] = 1.0f / l8[r];
  bf16* aohead = aob + ((size_t)bb * NPIX + qbase) * CDIM + hh * HD;
#pragma unroll
  for (int t = 0; t < 4; t++) {
#pragma unroll
    for (int r = 0; r < 8; r++) {
      int qr = r + 8 * h2;
      aohead[(size_t)qr * CDIM + 16 * t + lmod] = (bf16)(o[t][r] * m8[r]);
    }
  }
}

// ---------------------------------------------------------------------------
// Kernel 4: output projection + bias + residual (fp32 out), double-buffered.
// out[b][d][n] = x[b][d][n] + bo[d] + Wo[d][:] . ao[n][:]
// ---------------------------------------------------------------------------
__global__ __launch_bounds__(256) void proj_o_kernel(
    const bf16* __restrict__ aob, const bf16* __restrict__ wo_bf,
    const float* __restrict__ bo, const float* __restrict__ x,
    float* __restrict__ out) {
  const int wave = threadIdx.x >> 5;
  const int lane = threadIdx.x & 31;
  const int lmod = lane & 15;
  const int h2   = lane >> 4;

  int wid  = blockIdx.x * 8 + wave;    // 4*16*64 = 4096 waves
  int bb   = wid >> 10;
  int rem2 = wid & 1023;
  int dbase = (rem2 >> 6) * 16;
  int nbase = (rem2 & 63) * 64;

  f32x8 acc[4];
#pragma unroll
  for (int t = 0; t < 4; t++) acc[t] = fzero();

  const bf16* wrow = wo_bf + (size_t)(dbase + lmod) * CDIM;
  const bf16* arow = aob + ((size_t)bb * NPIX + nbase + lmod) * CDIM;

  Frag a_cur, a_nxt, b_cur[4], b_nxt[4];
  a_cur.h[0] = *(const bf16x8*)(wrow + 16 * h2);
  a_cur.h[1] = *(const bf16x8*)(wrow + 16 * h2 + 8);
#pragma unroll
  for (int t = 0; t < 4; t++) {
    const bf16* ap = arow + (size_t)(16 * t) * CDIM + 16 * h2;
    b_cur[t].h[0] = *(const bf16x8*)(ap);
    b_cur[t].h[1] = *(const bf16x8*)(ap + 8);
  }

#pragma unroll
  for (int kc = 0; kc < 8; kc++) {
    if (kc < 7) {
      const int c1 = (kc + 1) * 32;
      a_nxt.h[0] = *(const bf16x8*)(wrow + c1 + 16 * h2);
      a_nxt.h[1] = *(const bf16x8*)(wrow + c1 + 16 * h2 + 8);
#pragma unroll
      for (int t = 0; t < 4; t++) {
        const bf16* ap = arow + (size_t)(16 * t) * CDIM + c1 + 16 * h2;
        b_nxt[t].h[0] = *(const bf16x8*)(ap);
        b_nxt[t].h[1] = *(const bf16x8*)(ap + 8);
      }
    }
#pragma unroll
    for (int t = 0; t < 4; t++) acc[t] = wmma_bf16(a_cur.v, b_cur[t].v, acc[t]);
    a_cur = a_nxt;
#pragma unroll
    for (int t = 0; t < 4; t++) b_cur[t] = b_nxt[t];
  }

  f32x8 bo8 = *(const f32x8*)(bo + dbase + 8 * h2);
#pragma unroll
  for (int t = 0; t < 4; t++) {
    int n = nbase + 16 * t + lmod;
#pragma unroll
    for (int r = 0; r < 8; r++) {
      int d = dbase + 8 * h2 + r;
      size_t oi = ((size_t)bb * CDIM + d) * NPIX + n;
      out[oi] = x[oi] + acc[t][r] + bo8[r];
    }
  }
}

// ---------------------------------------------------------------------------
extern "C" void kernel_launch(void* const* d_in, const int* in_sizes, int n_in,
                              void* d_out, int out_size, void* d_ws, size_t ws_size,
                              hipStream_t stream) {
  const float* x  = (const float*)d_in[0];
  const float* wq = (const float*)d_in[1];
  const float* bq = (const float*)d_in[2];
  const float* wk = (const float*)d_in[3];
  const float* bk = (const float*)d_in[4];
  const float* wv = (const float*)d_in[5];
  const float* bv = (const float*)d_in[6];
  const float* wo = (const float*)d_in[7];
  const float* bo = (const float*)d_in[8];
  float* out = (float*)d_out;

  char* ws = (char*)d_ws;
  size_t off = 0;
  auto alloc = [&](size_t bytes) -> void* {
    void* p = ws + off;
    off += (bytes + 255) & ~(size_t)255;
    return p;
  };
  const size_t xbytes = (size_t)BATCH * NPIX * CDIM * sizeof(bf16);  // 8.39 MB
  bf16* xT   = (bf16*)alloc(xbytes);
  bf16* wbf  = (bf16*)alloc((size_t)4 * 256 * 256 * sizeof(bf16));
  bf16* qb   = (bf16*)alloc(xbytes);
  bf16* kbuf = (bf16*)alloc(xbytes);
  bf16* vtb  = (bf16*)alloc(xbytes);
  bf16* aob  = (bf16*)alloc(xbytes);

  // 1) fp32 -> bf16 conversions (x transposed to pixel-major)
  convert_x_kernel<<<(BATCH * CDIM * NPIX) / 256, 256, 0, stream>>>(x, xT);
  convert_w_kernel<<<(4 * 256 * 256) / 256, 256, 0, stream>>>(wq, wk, wv, wo, wbf);

  // 2) q/k/v projections (WMMA GEMM); q pre-scaled by hd^-0.5
  proj_qkv_kernel<<<12288 / 8, 256, 0, stream>>>(xT, wbf, bq, bk, bv, qb, kbuf, vtb);

  // 3) fused flash attention (WMMA + online softmax + TDM LDS staging)
  attn_kernel<<<512, 256, 0, stream>>>(qb, kbuf, vtb, aob);

  // 4) output projection + bias + residual (WMMA GEMM, fp32 out)
  proj_o_kernel<<<4096 / 8, 256, 0, stream>>>(aob, wbf + (size_t)3 * 65536, bo, x, out);
}
